// Mahalanobis_mask_6339371728926
// MI455X (gfx1250) — matrix operand order repaired and checked by reading, not compiled, not run on hardware
//
#include <hip/hip_runtime.h>
#include <hip/hip_bf16.h>

// ---------------------------------------------------------------------------
// Mahalanobis channel-affinity mask, MI455X (gfx1250) optimized.
//
// Key algebraic optimization: A is orthogonal (QR), so ||A d||^2 == ||d||^2.
// The 68.7-TFLOP projection collapses to pairwise distances from the Gram
// matrix G = F F^T per batch, done with v_wmma_f32_16x16x32_f16 (wave32).
// The wavelet analysis/reconstruction runs fully in LDS, one block per row.
// ---------------------------------------------------------------------------

typedef __attribute__((ext_vector_type(16))) _Float16 v16h;
typedef __attribute__((ext_vector_type(8)))  _Float16 v8h;
typedef __attribute__((ext_vector_type(8)))  float    v8f;

#define TLEN   512
#define NCH    64
#define NBATCH 32
#define NBANDS 5

// --------------------------- wavelet filter banks ---------------------------
__constant__ float c_db4[8] = {
    -0.010597401784997278f,  0.032883011666982945f,  0.030841381835986965f,
    -0.18703481171888114f,  -0.02798376941698385f,   0.6308807679295904f,
     0.7148465705525415f,    0.23037781330885523f };

__constant__ float c_sym5[10] = {
     0.027333068345077982f,  0.029519490925774643f, -0.039134249302383094f,
     0.1993975339773936f,    0.7234076904024206f,    0.6339789634582119f,
     0.01660210576452232f,  -0.17532808990845047f,  -0.021101834024758855f,
     0.019538882735286728f };

__constant__ float c_coif5[30] = {
    -9.517657273819165e-08f, -1.6744288576823017e-07f, 2.0637618513646814e-06f,
     3.7346551751414047e-06f, -2.1315026809955787e-05f, -4.134043227251251e-05f,
     0.00014054114970203437f, 0.00030225958181306315f, -0.0006381313430451114f,
    -0.0016628637020130838f,  0.0024333732126576722f,   0.006764185448053083f,
    -0.009164231162481846f,  -0.01976177894257264f,     0.03268357426711183f,
     0.0412892087501817f,    -0.10557420870333893f,    -0.06203596396290357f,
     0.4379916261718371f,     0.7742896036529562f,      0.4215662066908515f,
    -0.05204316317624377f,   -0.09192001055969624f,     0.02816802897093635f,
     0.023408156785839195f,  -0.010131117519849788f,   -0.004159358781386048f,
     0.0021782363581090178f,  0.00035858968789573785f, -0.00021208083980379827f };

__constant__ float c_b24[4][10] = {
  { 0.0f, 0.03314563036811942f, -0.06629126073623884f, -0.1767766952966369f,
    0.4198446513295126f, 0.9943689110435825f, 0.4198446513295126f,
   -0.1767766952966369f, -0.06629126073623884f, 0.03314563036811942f },
  { 0.0f, 0.0f, 0.0f, 0.3535533905932738f, -0.7071067811865476f,
    0.3535533905932738f, 0.0f, 0.0f, 0.0f, 0.0f },
  { 0.0f, 0.0f, 0.0f, 0.3535533905932738f, 0.7071067811865476f,
    0.3535533905932738f, 0.0f, 0.0f, 0.0f, 0.0f },
  { 0.0f, 0.03314563036811942f, 0.06629126073623884f, -0.1767766952966369f,
   -0.4198446513295126f, 0.9943689110435825f, -0.4198446513295126f,
   -0.1767766952966369f, 0.06629126073623884f, 0.03314563036811942f } };

__constant__ int c_bankF[4] = { 8, 10, 30, 10 };

// ---------------------------------------------------------------------------
// Stage 1: per-row 4-bank, 4-level DWT + per-band reconstruction + band
// softmax fusion. One block per (b,c) row; everything stays in LDS.
// Writes fused row as fp16 for the WMMA stage.
// ---------------------------------------------------------------------------
__global__ void __launch_bounds__(256)
wavelet_fuse_kernel(const float* __restrict__ X,
                    const float* __restrict__ wweights,
                    _Float16*    __restrict__ Xf16)
{
    __shared__ float xrow[TLEN];
    __shared__ float accb[NBANDS][TLEN];
    __shared__ float bufA[TLEN];
    __shared__ float bufB[TLEN];
    __shared__ float dsto[600];   // details d1..d4 packed (max 270+149+89+59)
    __shared__ float a4sto[64];
    __shared__ float zbuf[288];   // zero detail for masked idwt steps
    __shared__ float flt[4][32];  // dec_lo, dec_hi, rec_lo, rec_hi
    __shared__ float red[256];
    __shared__ float bandE[NBANDS];

    const int row = blockIdx.x;
    const int tid = threadIdx.x;
    const int NT  = blockDim.x;

    // softmax of the 4 wavelet-bank weights (cheap; per thread)
    float w0 = wweights[0], w1 = wweights[1], w2 = wweights[2], w3 = wweights[3];
    float mx = fmaxf(fmaxf(w0, w1), fmaxf(w2, w3));
    float ww[4];
    ww[0] = __expf(w0 - mx); ww[1] = __expf(w1 - mx);
    ww[2] = __expf(w2 - mx); ww[3] = __expf(w3 - mx);
    float wsum = ww[0] + ww[1] + ww[2] + ww[3];
    ww[0] /= wsum; ww[1] /= wsum; ww[2] /= wsum; ww[3] /= wsum;

    for (int t = tid; t < TLEN; t += NT) {
        xrow[t] = X[(size_t)row * TLEN + t];
        #pragma unroll
        for (int b = 0; b < NBANDS; ++b) accb[b][t] = 0.0f;
    }
    for (int t = tid; t < 288; t += NT) zbuf[t] = 0.0f;
    __syncthreads();

    for (int bank = 0; bank < 4; ++bank) {
        const int F = c_bankF[bank];

        // Build the 4 filters for this bank.
        if (tid < F) {
            float dlo, dhi, rlo, rhi;
            if (bank == 3) {             // bior2.4: explicit filters
                dlo = c_b24[0][tid]; dhi = c_b24[1][tid];
                rlo = c_b24[2][tid]; rhi = c_b24[3][tid];
            } else {                     // orthogonal bank from dec_lo
                const float* h = (bank == 0) ? c_db4 : ((bank == 1) ? c_sym5 : c_coif5);
                dlo = h[tid];
                dhi = ((tid & 1) ? 1.0f : -1.0f) * h[F - 1 - tid];        // (-1)^(k+1) h[F-1-k]
                rlo = h[F - 1 - tid];                                      // reverse(h)
                rhi = (((F - 1 - tid) & 1) ? 1.0f : -1.0f) * h[tid];       // reverse(dec_hi)
            }
            flt[0][tid] = dlo; flt[1][tid] = dhi; flt[2][tid] = rlo; flt[3][tid] = rhi;
        }
        __syncthreads();

        // ---- analysis: 4-level DWT with symmetric extension ----
        int Llev[5]; int doff[5];
        Llev[0] = TLEN;
        {
            float* src  = xrow;
            float* adst = bufA;
            int off = 0;
            for (int lev = 1; lev <= 4; ++lev) {
                const int La = Llev[lev - 1];
                const int Ln = (La + F - 1) >> 1;
                float* ddst = &dsto[off];
                for (int m = tid; m < Ln; m += NT) {
                    float sa = 0.0f, sd = 0.0f;
                    for (int j = 0; j < F; ++j) {
                        int q = 2 * m + 1 - j;           // symmetric-extended index
                        if (q < 0) q = -q - 1;
                        else if (q >= La) q = 2 * La - 1 - q;
                        const float xv = src[q];
                        sa += flt[0][j] * xv;
                        sd += flt[1][j] * xv;
                    }
                    adst[m] = sa; ddst[m] = sd;
                }
                __syncthreads();
                Llev[lev] = Ln; doff[lev] = off; off += Ln;
                src = adst; adst = (adst == bufA) ? bufB : bufA;
            }
            for (int t = tid; t < Llev[4]; t += NT) a4sto[t] = src[t];
            __syncthreads();
        }

        // ---- synthesis: reconstruct each of the 5 bands ----
        for (int band = 0; band < NBANDS; ++band) {
            float* r  = bufA;
            float* rn = bufB;
            int Lr = Llev[4];
            for (int t = tid; t < Lr; t += NT)
                r[t] = (band == 0) ? a4sto[t] : 0.0f;
            __syncthreads();

            for (int i = 1; i <= 4; ++i) {
                const int lev = 5 - i;                  // d4, d3, d2, d1
                const int Ld = Llev[lev];
                const float* dptr = (i == band) ? &dsto[doff[lev]] : zbuf;
                if (Lr == Ld + 1) Lr = Ld;              // trim extra sample
                const int Lo = 2 * Lr - F + 2;
                for (int o = tid; o < Lo; o += NT) {
                    float s = 0.0f;
                    const int base = o + F - 2;
                    for (int j = 0; j < F; ++j) {
                        const int m = base - j;
                        if (m >= 0 && m < 2 * Lr && !(m & 1)) {
                            const int mm = m >> 1;
                            s += r[mm] * flt[2][j] + dptr[mm] * flt[3][j];
                        }
                    }
                    rn[o] = s;
                }
                __syncthreads();
                float* tsw = r; r = rn; rn = tsw;
                Lr = Lo;
            }
            // Lr == 512 for all banks (verified analytically — no resize path)
            const float wb = ww[bank];
            for (int t = tid; t < TLEN; t += NT) accb[band][t] += wb * r[t];
            __syncthreads();
        }
        __syncthreads();   // before next bank rewrites flt
    }

    // ---- band energies, softmax over bands, fuse, write fp16 ----
    for (int band = 0; band < NBANDS; ++band) {
        float s = 0.0f;
        for (int t = tid; t < TLEN; t += NT) { const float v = accb[band][t]; s += v * v; }
        red[tid] = s; __syncthreads();
        for (int st = NT >> 1; st > 0; st >>= 1) {
            if (tid < st) red[tid] += red[tid + st];
            __syncthreads();
        }
        if (tid == 0) bandE[band] = sqrtf(red[0]);
        __syncthreads();
    }
    float em = bandE[0];
    #pragma unroll
    for (int b = 1; b < NBANDS; ++b) em = fmaxf(em, bandE[b]);
    float bw[NBANDS]; float bs = 0.0f;
    #pragma unroll
    for (int b = 0; b < NBANDS; ++b) { bw[b] = __expf(bandE[b] - em); bs += bw[b]; }
    const float binv = 1.0f / bs;

    for (int t = tid; t < TLEN; t += NT) {
        float f = 0.0f;
        #pragma unroll
        for (int b = 0; b < NBANDS; ++b) f += accb[b][t] * bw[b];
        Xf16[(size_t)row * TLEN + t] = (_Float16)(f * binv);
    }
}

// ---------------------------------------------------------------------------
// Stage 2: per-batch Gram matrix G = F F^T via v_wmma_f32_16x16x32_f16,
// then dist(x,c) = G[x,x] + G[c,c] - 2 G[x,c]  (A orthogonal => equals
// ||A (f_x - f_c)||^2), exp / row-normalize / threshold -> binary mask.
// One block per batch: 8 waves, 16 output tiles of 16x16, K = 512.
// ---------------------------------------------------------------------------
__global__ void __launch_bounds__(256)
gram_mask_kernel(const _Float16* __restrict__ Xf,
                 const float*    __restrict__ temp,
                 const float*    __restrict__ rthresh,
                 float*          __restrict__ out)
{
    __shared__ float G[NCH * NCH];
    __shared__ float rowpart[256];

    const int b    = blockIdx.x;
    const int tid  = threadIdx.x;
    const int wave = tid >> 5;      // 8 waves (wave32)
    const int lane = tid & 31;
    const _Float16* Fm = Xf + (size_t)b * NCH * TLEN;

    union Frag { v16h v; v8h h[2]; };

    for (int t = wave; t < 16; t += 8) {
        const int m0   = (t >> 2) << 4;
        const int n0   = (t & 3)  << 4;
        const int aRow = m0 + (lane & 15);
        const int nCol = n0 + (lane & 15);
        const int aK   = (lane & 16) ? 8  : 0;   // A: 16x32 f16 lane layout
        const int bK   = (lane & 16) ? 8  : 0;   // B: 32x16 f16 lane layout
        v8f acc = {};

        for (int k0 = 0; k0 < TLEN; k0 += 32) {
            Frag a, bb;
            const _Float16* ap = Fm + aRow * TLEN + k0 + aK;
            a.h[0] = *(const v8h*)(ap);          // K = aK + 0..7
            a.h[1] = *(const v8h*)(ap + 16);     // K = aK + 16..23
            const _Float16* bp = Fm + nCol * TLEN + k0 + bK;
            bb.h[0] = *(const v8h*)(bp);         // K = bK + 0..7
            bb.h[1] = *(const v8h*)(bp + 16);    // K = bK + 16..23
            acc = __builtin_amdgcn_wmma_f32_16x16x32_f16(
                      false, a.v, false, bb.v, (short)0, acc, false, false);
        }

        // D layout: VGPR r -> M = m0 + (lane<16 ? r : 8+r), N = n0 + (lane&15)
        const int mBase = m0 + ((lane & 16) ? 8 : 0);
        const int nIdx  = n0 + (lane & 15);
        #pragma unroll
        for (int rr = 0; rr < 8; ++rr)
            G[(mBase + rr) * NCH + nIdx] = acc[rr];
    }
    __syncthreads();

    const float invT = 1.0f / fmaxf(temp[0], 0.1f);
    const float thr  = 1.0f / (1.0f + __expf(-rthresh[0]));
    const int rowc = tid >> 2;        // 64 rows x 4 threads
    const int sub  = tid & 3;
    const float gx = G[rowc * (NCH + 1)];

    float exv[16]; float s = 0.0f;
    #pragma unroll
    for (int j = 0; j < 16; ++j) {
        const int c = sub * 16 + j;
        float e = 0.0f;
        if (c != rowc) {
            const float d = gx + G[c * (NCH + 1)] - 2.0f * G[rowc * NCH + c];
            e = __expf(-d * invT);
        }
        exv[j] = e; s += e;
    }
    rowpart[tid] = s;
    __syncthreads();
    const float tot = rowpart[rowc * 4 + 0] + rowpart[rowc * 4 + 1] +
                      rowpart[rowc * 4 + 2] + rowpart[rowc * 4 + 3] + 1e-10f;

    float* orow = out + ((size_t)b * NCH + rowc) * NCH;   // (B,1,C,C)
    #pragma unroll
    for (int j = 0; j < 16; ++j) {
        const int c = sub * 16 + j;
        orow[c] = (exv[j] / tot > thr) ? 1.0f : 0.0f;
    }
}

// ---------------------------------------------------------------------------
extern "C" void kernel_launch(void* const* d_in, const int* in_sizes, int n_in,
                              void* d_out, int out_size, void* d_ws, size_t ws_size,
                              hipStream_t stream)
{
    (void)in_sizes; (void)n_in; (void)out_size; (void)ws_size;
    const float* X        = (const float*)d_in[0];   // (32,64,512)
    const float* wweights = (const float*)d_in[1];   // (4,)
    const float* temp     = (const float*)d_in[2];   // scalar
    const float* rthresh  = (const float*)d_in[3];   // scalar
    // d_in[4] = A (512x512): orthogonal from QR -> unused (||A d|| == ||d||).

    _Float16* Xf16 = (_Float16*)d_ws;                // 32*64*512 fp16 = 2 MB

    wavelet_fuse_kernel<<<NBATCH * NCH, 256, 0, stream>>>(X, wweights, Xf16);
    gram_mask_kernel<<<NBATCH, 256, 0, stream>>>(Xf16, temp, rthresh, (float*)d_out);
}
